// Composer_17119739642120
// MI455X (gfx1250) — matrix operand order repaired and checked by reading, not compiled, not run on hardware
//
#include <hip/hip_runtime.h>
#include <hip/hip_bf16.h>

// ---------------------------------------------------------------------------
// MI455X (gfx1250) implementation.
// Both bilinears are outer-product GEMMs on v_wmma_f32_16x16x32_bf16 (f32
// accumulate). bf16 weights are cast once (L2-resident). The bf16 A-panel
// (tok (x) dep / tok (x) tde) is built cooperatively in LDS per 256-wide
// K-chunk so each product is computed once per workgroup; the WMMA consumer
// loop is fully unrolled and reads A via 16B ds loads with zero pack VALU.
// Head-scatter reduction uses global f32 atomics into a ping-pong buffer.
// ---------------------------------------------------------------------------

typedef __attribute__((ext_vector_type(16))) __bf16 v16bf;
typedef __attribute__((ext_vector_type(8)))  float  v8f;

#define NB   16
#define NS   128
#define ND   128
#define NE   64
#define NT   128
#define NTOK (NB * NS)          // 2048
#define K1   (ND * NE)          // 8192
#define K2   (ND * NT)          // 16384
#define KC   256                // K-chunk staged in the LDS A-panel

struct UQ2 { uint4 a, b; };     // 32 bytes -> v16bf

__device__ __forceinline__ unsigned short f2bf_rne(float f) {
  unsigned int u = __builtin_bit_cast(unsigned int, f);
  u += 0x7FFFu + ((u >> 16) & 1u);
  return (unsigned short)(u >> 16);
}
__device__ __forceinline__ unsigned short f2bf_trunc(float f) {
  return (unsigned short)(__builtin_bit_cast(unsigned int, f) >> 16);
}

// ---------------- small prep kernels ----------------------------------------

__global__ void gather_tok_kernel(const float* __restrict__ table,
                                  const int* __restrict__ tokens,
                                  float* __restrict__ tok) {
  int i = blockIdx.x * blockDim.x + threadIdx.x;
  if (i < NTOK * ND) {
    int n = i >> 7, d = i & 127;
    tok[i] = table[(size_t)tokens[n] * ND + d];
  }
}

__global__ void gather_dep_kernel(const float* __restrict__ table,
                                  const int* __restrict__ types,
                                  float* __restrict__ dep) {
  int i = blockIdx.x * blockDim.x + threadIdx.x;
  if (i < NTOK * NE) {
    int n = i >> 6, e = i & 63;
    dep[i] = table[(size_t)types[n] * NE + e];
  }
}

__global__ void cast_weights_kernel(const float* __restrict__ W1,
                                    const float* __restrict__ W2,
                                    unsigned short* __restrict__ W1b,
                                    unsigned short* __restrict__ W2b) {
  int i = blockIdx.x * blockDim.x + threadIdx.x;
  const int n1 = NT * K1;                  // 1,048,576
  const int n2 = ND * K2;                  // 2,097,152
  if (i < n1)            W1b[i]      = f2bf_rne(W1[i]);
  else if (i < n1 + n2)  W2b[i - n1] = f2bf_rne(W2[i - n1]);
}

__global__ void prep_kernel(const float* __restrict__ b2,
                            const float* __restrict__ wr,
                            const float* __restrict__ br,
                            float* __restrict__ cbg,
                            float* __restrict__ base) {
  __shared__ float red[NS];
  int t = threadIdx.x;                      // 128 threads
  red[t] = wr[t];
  __syncthreads();
  for (int off = 64; off > 0; off >>= 1) {
    if (t < off) red[t] += red[t + off];
    __syncthreads();
  }
  float sw = red[0];
  float cg = tanhf(b2[t]);
  cbg[t]  = cg;
  base[t] = sw * cg + br[0];
}

__global__ void init_next_kernel(float* __restrict__ nxt,
                                 const float* __restrict__ base) {
  int i = blockIdx.x * blockDim.x + threadIdx.x;
  if (i < NTOK * ND) nxt[i] = base[i & 127];
}

__global__ void mask_out_kernel(const float* __restrict__ cur,
                                const int* __restrict__ heads,
                                float* __restrict__ out) {
  int i = blockIdx.x * blockDim.x + threadIdx.x;
  if (i < NTOK * ND) out[i] = (heads[i >> 7] == 0) ? cur[i] : 0.0f;
}

// ---------------- fused block kernel ----------------------------------------
// grid: 64 workgroups x 256 threads (8 wave32). WG g owns tokens
// [g*32, g*32+32). Wave w: M-tile (w&1) of 16 rows, 32-col slab (w>>1)*32.

__global__ __launch_bounds__(256)
void block_kernel(const float* __restrict__ tok_cur,
                  float* __restrict__ tok_next,
                  const float* __restrict__ dep_f,
                  const unsigned short* __restrict__ W1b,
                  const unsigned short* __restrict__ W2b,
                  const float* __restrict__ b1,
                  const float* __restrict__ b2,
                  const float* __restrict__ wr,
                  const float* __restrict__ cbg,
                  const int* __restrict__ heads) {
  __shared__ float tok_lds[32][132];            // fp32 activations (padded)
  __shared__ float dep_lds[32][68];
  __shared__ float tde_lds[32][132];
  __shared__ unsigned short apan[32][KC + 16];  // shared bf16 A-panel

  const int tid  = threadIdx.x;
  const int g    = blockIdx.x;
  const int row0 = g * 32;

  // Stage this group's activations into LDS.
  for (int i = tid; i < 32 * ND; i += 256) {
    int r = i >> 7, c = i & 127;
    tok_lds[r][c] = tok_cur[(size_t)(row0 + r) * ND + c];
  }
  for (int i = tid; i < 32 * NE; i += 256) {
    int r = i >> 6, c = i & 63;
    dep_lds[r][c] = dep_f[(size_t)(row0 + r) * NE + c];
  }

  const int lane = tid & 31;
  const int w    = tid >> 5;
  const int m    = w & 1;              // M-tile
  const int c0   = (w >> 1) * 32;      // 32 output columns (2 N-tiles)
  const int ln   = lane & 15;
  const int lh   = lane >> 4;
  const int arow = m * 16 + ln;        // A-matrix row for this lane
  const int pr   = tid >> 3;           // panel-gen row (32 rows, 8 thr/row)
  const int pk0  = (tid & 7) * 32;     // panel-gen k-span within chunk

  // Warm the L2/WGP$ path for the W2 rows this wave will stream.
  __builtin_prefetch(W2b + (size_t)(c0 + ln) * K2, 0, 1);
  __builtin_prefetch(W2b + (size_t)(c0 + 16 + ln) * K2, 0, 1);

  __syncthreads();

  // ---------------- GEMM1: tde = tanh((tok (x) dep) * W1^T + b1) -----------
  v8f acc1a = {0.f, 0.f, 0.f, 0.f, 0.f, 0.f, 0.f, 0.f};
  v8f acc1b = {0.f, 0.f, 0.f, 0.f, 0.f, 0.f, 0.f, 0.f};
  {
    const unsigned short* bw0 = W1b + (size_t)(c0 + ln) * K1;
    const unsigned short* bw1 = W1b + (size_t)(c0 + 16 + ln) * K1;
    for (int kc = 0; kc < K1; kc += KC) {
      // Cooperative panel generation: each bf16 product computed once per WG.
      {
        const int   kg = kc + pk0;       // 32-wide span stays inside one d
        const int   d  = kg >> 6;
        const int   e0 = kg & 63;
        const float tv = tok_lds[pr][d];
#pragma unroll
        for (int i = 0; i < 32; ++i)
          apan[pr][pk0 + i] = f2bf_trunc(tv * dep_lds[pr][e0 + i]);
      }
      __syncthreads();
#pragma unroll
      for (int s = 0; s < KC / 32; ++s) {
        const int kb = s * 32;
        const unsigned short* ap = &apan[arow][kb + lh * 16];
        UQ2 qa = {*(const uint4*)ap, *(const uint4*)(ap + 8)};
        v16bf av = __builtin_bit_cast(v16bf, qa);

        const uint4* p0 = (const uint4*)(bw0 + kc + kb + lh * 16);
        const uint4* p1 = (const uint4*)(bw1 + kc + kb + lh * 16);
        UQ2 q0 = {p0[0], p0[1]};
        UQ2 q1 = {p1[0], p1[1]};
        v16bf bv0 = __builtin_bit_cast(v16bf, q0);
        v16bf bv1 = __builtin_bit_cast(v16bf, q1);

        acc1a = __builtin_amdgcn_wmma_f32_16x16x32_bf16(
            false, av, false, bv0, (short)0, acc1a, false, false);
        acc1b = __builtin_amdgcn_wmma_f32_16x16x32_bf16(
            false, av, false, bv1, (short)0, acc1b, false, false);
      }
      __syncthreads();
    }
  }
  // tanh epilogue -> tde in LDS.
#pragma unroll
  for (int nt = 0; nt < 2; ++nt) {
    v8f a = nt ? acc1b : acc1a;
    int col = c0 + nt * 16 + ln;
    float bb = b1[col];
#pragma unroll
    for (int r = 0; r < 8; ++r) {
      tde_lds[m * 16 + r + 8 * lh][col] = tanhf(a[r] + bb);
    }
  }
  __syncthreads();

  // ---------------- GEMM2: c = tanh((tok (x) tde) * W2^T + b2) -------------
  v8f acc2a = {0.f, 0.f, 0.f, 0.f, 0.f, 0.f, 0.f, 0.f};
  v8f acc2b = {0.f, 0.f, 0.f, 0.f, 0.f, 0.f, 0.f, 0.f};
  {
    const unsigned short* bw0 = W2b + (size_t)(c0 + ln) * K2;
    const unsigned short* bw1 = W2b + (size_t)(c0 + 16 + ln) * K2;
    for (int kc = 0; kc < K2; kc += KC) {
      {
        const int   kg = kc + pk0;       // 32-wide span stays inside one d
        const int   d  = kg >> 7;
        const int   t0 = kg & 127;
        const float tv = tok_lds[pr][d];
#pragma unroll
        for (int i = 0; i < 32; ++i)
          apan[pr][pk0 + i] = f2bf_trunc(tv * tde_lds[pr][t0 + i]);
      }
      __syncthreads();
#pragma unroll
      for (int s = 0; s < KC / 32; ++s) {
        const int kb = s * 32;
        const unsigned short* ap = &apan[arow][kb + lh * 16];
        UQ2 qa = {*(const uint4*)ap, *(const uint4*)(ap + 8)};
        v16bf av = __builtin_bit_cast(v16bf, qa);

        const uint4* p0 = (const uint4*)(bw0 + kc + kb + lh * 16);
        const uint4* p1 = (const uint4*)(bw1 + kc + kb + lh * 16);
        UQ2 q0 = {p0[0], p0[1]};
        UQ2 q1 = {p1[0], p1[1]};
        v16bf bv0 = __builtin_bit_cast(v16bf, q0);
        v16bf bv1 = __builtin_bit_cast(v16bf, q1);

        acc2a = __builtin_amdgcn_wmma_f32_16x16x32_bf16(
            false, av, false, bv0, (short)0, acc2a, false, false);
        acc2b = __builtin_amdgcn_wmma_f32_16x16x32_bf16(
            false, av, false, bv1, (short)0, acc2b, false, false);
      }
      __syncthreads();
    }
  }
  // tanh + head-scatter epilogue: tok_next[b, heads[n], :] += wr[s]*(c - cbg)
#pragma unroll
  for (int nt = 0; nt < 2; ++nt) {
    v8f a = nt ? acc2b : acc2a;
    int col = c0 + nt * 16 + ln;
    float bb = b2[col];
    float cg = cbg[col];
#pragma unroll
    for (int r = 0; r < 8; ++r) {
      int nl = m * 16 + r + 8 * lh;
      int n  = row0 + nl;                 // global token index = b*S + s
      float v     = tanhf(a[r] + bb);
      float delta = wr[n & 127] * (v - cg);
      int head = heads[n];
      int b    = n >> 7;
      atomicAdd(&tok_next[((size_t)(b * NS + head)) * ND + col], delta);
    }
  }
}

// ---------------- host-side launch ------------------------------------------

extern "C" void kernel_launch(void* const* d_in, const int* in_sizes, int n_in,
                              void* d_out, int out_size, void* d_ws, size_t ws_size,
                              hipStream_t stream) {
  (void)in_sizes; (void)n_in; (void)out_size; (void)ws_size;
  const float* token_table = (const float*)d_in[0];
  const float* dep_table   = (const float*)d_in[1];
  const float* W1          = (const float*)d_in[2];
  const float* b1          = (const float*)d_in[3];
  const float* W2          = (const float*)d_in[4];
  const float* b2          = (const float*)d_in[5];
  const float* wr          = (const float*)d_in[6];
  const float* br          = (const float*)d_in[7];
  const int*   tokens      = (const int*)d_in[8];
  const int*   dep_types   = (const int*)d_in[9];
  const int*   dep_heads   = (const int*)d_in[10];
  float* out = (float*)d_out;

  char* wsp = (char*)d_ws;
  auto carve = [&](size_t bytes) -> char* {
    char* p = wsp;
    wsp += (bytes + 255) & ~(size_t)255;
    return p;
  };
  float*          tokA = (float*)carve((size_t)NTOK * ND * 4);
  float*          tokB = (float*)carve((size_t)NTOK * ND * 4);
  float*          depf = (float*)carve((size_t)NTOK * NE * 4);
  unsigned short* W1b  = (unsigned short*)carve((size_t)NT * K1 * 2);
  unsigned short* W2b  = (unsigned short*)carve((size_t)ND * K2 * 2);
  float*          cbg  = (float*)carve(ND * 4);
  float*          base = (float*)carve(ND * 4);

  gather_tok_kernel<<<(NTOK * ND) / 256, 256, 0, stream>>>(token_table, tokens, tokA);
  gather_dep_kernel<<<(NTOK * NE) / 256, 256, 0, stream>>>(dep_table, dep_types, depf);
  cast_weights_kernel<<<(NT * K1 + ND * K2) / 256, 256, 0, stream>>>(W1, W2, W1b, W2b);
  prep_kernel<<<1, 128, 0, stream>>>(b2, wr, br, cbg, base);

  const float* cur = tokA;
  float*       nxt = tokB;
  for (int it = 0; it < 2; ++it) {
    init_next_kernel<<<(NTOK * ND) / 256, 256, 0, stream>>>(nxt, base);
    block_kernel<<<64, 256, 0, stream>>>(cur, nxt, depf, W1b, W2b,
                                         b1, b2, wr, cbg, dep_heads);
    const float* t = nxt;
    nxt = (float*)cur;
    cur = t;
  }
  mask_out_kernel<<<(NTOK * ND) / 256, 256, 0, stream>>>(cur, dep_heads, out);
}
